// MultiHeadAttention_74251394613805
// MI455X (gfx1250) — matrix operand order repaired and checked by reading, not compiled, not run on hardware
//
#include <hip/hip_runtime.h>

#define BATCH  2
#define S_LEN  2048
#define DMODEL 1024
#define NHEAD  16
#define HDIM   64

// LDS leading dims (elements), padded: 16B-aligned row strides, no 16-lane
// bank collisions (stride in dwords: 36 / 20 / 20 -> first repeat at lane 16).
#define KLD 72
#define VLD 40
#define PLD 40

typedef __attribute__((ext_vector_type(16))) __bf16 v16bf;
typedef __attribute__((ext_vector_type(8)))  __bf16 v8bf;
typedef __attribute__((ext_vector_type(8)))  float  v8f;

__device__ __forceinline__ v8f wmma_bf16(v16bf a, v16bf b, v8f c) {
  // (neg_a, A, neg_b, B, c_mod, C, reuse_a, reuse_b)
  return __builtin_amdgcn_wmma_f32_16x16x32_bf16(false, a, false, b, (short)0, c,
                                                 false, false);
}

// A-matrix fragment 16x32 bf16 from row-major source (leading dim = lda elems).
// lane<16: halves 0..7 = K0..7, 8..15 = K16..23; lane>=16: +8. Row M = lane&15.
__device__ __forceinline__ v16bf load_a_frag(const __bf16* __restrict__ base,
                                             int lda, int lane) {
  const int row = lane & 15;
  const int sel = lane >> 4;
  const __bf16* p = base + (size_t)row * lda + sel * 8;
  v8bf lo = *(const v8bf*)(p);
  v8bf hi = *(const v8bf*)(p + 16);
  v16bf f;
#pragma unroll
  for (int i = 0; i < 8; ++i) { f[i] = lo[i]; f[i + 8] = hi[i]; }
  return f;
}

// B-matrix fragment 32x16 bf16 from K-contiguous source: element (k,n) at
// base[n*ldn + k]. Lane layout: n = lane&15, i<8: K = (lane>>4)*8 + i, i>=8: K = 16 + (lane>>4)*8 + (i-8).
__device__ __forceinline__ v16bf load_b_frag(const __bf16* __restrict__ base,
                                             int ldn, int lane) {
  const int col = lane & 15;
  const int sel = lane >> 4;
  const __bf16* p = base + (size_t)col * ldn + sel * 8;
  v16bf f;
#pragma unroll
  for (int i = 0; i < 8; ++i) { f[i] = p[i]; f[i + 8] = p[16 + i]; }
  return f;
}

__global__ __launch_bounds__(256) void cast_f32_bf16(const float* __restrict__ in,
                                                     __bf16* __restrict__ out, int n) {
  int i = blockIdx.x * blockDim.x + threadIdx.x;
  if (i < n) out[i] = (__bf16)in[i];
}

// C(4096 x 1024) = A(row-major, lda=1024, bf16) x W^T  (W is N x K row-major).
// 8 waves/block; wave tile 64x32 (4x2 WMMA accs); block tile 128x128.
// MODE 0: bf16 out, head-split [B,H,S,HD]
// MODE 1: bf16 out, head-split transposed [B,H,HD,S]
// MODE 2: f32 out, plain [M,N]
template <int MODE>
__global__ __launch_bounds__(256, 1) void gemm_xWT(const __bf16* __restrict__ A,
                                                   const __bf16* __restrict__ W,
                                                   void* __restrict__ out) {
  const int lane = threadIdx.x & 31;
  const int wave = threadIdx.x >> 5;
  const int m0 = blockIdx.y * 128 + (wave >> 2) * 64;
  const int n0 = blockIdx.x * 128 + (wave & 3) * 32;

  v8f acc[4][2];
#pragma unroll
  for (int i = 0; i < 4; ++i)
#pragma unroll
    for (int j = 0; j < 2; ++j) acc[i][j] = {};

  for (int k0 = 0; k0 < DMODEL; k0 += 32) {
    if (k0 + 32 < DMODEL) {
      __builtin_prefetch(A + (size_t)m0 * DMODEL + k0 + 32, 0, 1);
      __builtin_prefetch(W + (size_t)n0 * DMODEL + k0 + 32, 0, 1);
    }
    v16bf af[4], bfr[2];
#pragma unroll
    for (int i = 0; i < 4; ++i)
      af[i] = load_a_frag(A + (size_t)(m0 + i * 16) * DMODEL + k0, DMODEL, lane);
#pragma unroll
    for (int j = 0; j < 2; ++j)
      bfr[j] = load_b_frag(W + (size_t)(n0 + j * 16) * DMODEL + k0, DMODEL, lane);
#pragma unroll
    for (int i = 0; i < 4; ++i)
#pragma unroll
      for (int j = 0; j < 2; ++j)
        acc[i][j] = wmma_bf16(af[i], bfr[j], acc[i][j]);
  }

  const int colb = lane & 15;
  const int half = lane >> 4;
#pragma unroll
  for (int i = 0; i < 4; ++i)
#pragma unroll
    for (int j = 0; j < 2; ++j)
#pragma unroll
      for (int r = 0; r < 8; ++r) {
        const int row = m0 + i * 16 + r + half * 8;   // [0, 4096)
        const int col = n0 + j * 16 + colb;           // [0, 1024)
        const float vv = acc[i][j][r];
        if (MODE == 2) {
          ((float*)out)[(size_t)row * DMODEL + col] = vv;
        } else {
          const int b = row >> 11, s = row & (S_LEN - 1);
          const int h = col >> 6,  hd = col & (HDIM - 1);
          if (MODE == 0)
            ((__bf16*)out)[((size_t)(b * NHEAD + h) * S_LEN + s) * HDIM + hd] =
                (__bf16)vv;
          else
            ((__bf16*)out)[((size_t)(b * NHEAD + h) * HDIM + hd) * S_LEN + s] =
                (__bf16)vv;
        }
      }
}

// Causal flash attention. Qh/Kh: [B,H,S,HD] bf16; Vt: [B,H,HD,S] bf16.
// 8 waves/block; each wave owns 16 query rows. Block-uniform key loop with
// cooperative K/V staging in LDS; waves past their causal bound skip compute
// (wave-uniform branch: EXEC stays all-ones for WMMA) but keep barriers.
__global__ __launch_bounds__(256, 1) void flash_attn(const __bf16* __restrict__ Qh,
                                                     const __bf16* __restrict__ Kh,
                                                     const __bf16* __restrict__ Vt,
                                                     __bf16* __restrict__ attnOut) {
  __shared__ __align__(16) __bf16 ktile[32 * KLD];      // [t_local][hd]
  __shared__ __align__(16) __bf16 vtile[HDIM * VLD];    // [hd][t_local]
  __shared__ __align__(16) __bf16 pbuf[8][16 * PLD];    // per-wave P tile

  const int tid  = threadIdx.x;
  const int lane = tid & 31;
  const int wave = tid >> 5;
  const int bh = blockIdx.y;
  const int b  = bh >> 4;            // / NHEAD
  const int h  = bh & (NHEAD - 1);
  const int m0 = blockIdx.x * 128 + wave * 16;      // query row base in [0,S)
  const int mhi_blk = blockIdx.x * 128 + 127;       // block-uniform causal bound

  const __bf16* Q = Qh + (size_t)bh * S_LEN * HDIM;
  const __bf16* K = Kh + (size_t)bh * S_LEN * HDIM;
  const __bf16* V = Vt + (size_t)bh * HDIM * S_LEN;

  const v16bf qf0 = load_a_frag(Q + (size_t)m0 * HDIM + 0,  HDIM, lane);
  const v16bf qf1 = load_a_frag(Q + (size_t)m0 * HDIM + 32, HDIM, lane);

  v8f acc[4];
#pragma unroll
  for (int j = 0; j < 4; ++j) acc[j] = {};
  float mrow[8], lrow[8];
#pragma unroll
  for (int r = 0; r < 8; ++r) { mrow[r] = -1e30f; lrow[r] = 0.f; }

  const int colb = lane & 15;
  const int half = lane >> 4;

  for (int t0 = 0; t0 <= mhi_blk; t0 += 32) {
    __syncthreads();  // previous iteration's LDS reads complete
    // ---- stage K tile (32 x 64, contiguous rows): 8 thr/row x 16B ----
    {
      const int r = tid >> 3;
      const int c = (tid & 7) * 8;
      *(v8bf*)&ktile[r * KLD + c] = *(const v8bf*)(K + (size_t)(t0 + r) * HDIM + c);
    }
    // ---- stage V tile (64 hd-rows x 32 t): 4 thr/row x 16B ----
    {
      const int r = tid >> 2;
      const int c = (tid & 3) * 8;
      *(v8bf*)&vtile[r * VLD + c] = *(const v8bf*)(V + (size_t)r * S_LEN + t0 + c);
    }
    __syncthreads();

    if (t0 <= m0 + 15) {   // wave-uniform
      // ---- logits tile 16x32 = Q(16x64) @ K^T (from LDS) ----
      v8f lacc[2];
      lacc[0] = {}; lacc[1] = {};
#pragma unroll
      for (int j = 0; j < 2; ++j) {
        v16bf kf0 = load_b_frag(&ktile[(j * 16) * KLD + 0],  KLD, lane);
        v16bf kf1 = load_b_frag(&ktile[(j * 16) * KLD + 32], KLD, lane);
        lacc[j] = wmma_bf16(qf0, kf0, lacc[j]);
        lacc[j] = wmma_bf16(qf1, kf1, lacc[j]);
      }
      // ---- scale + causal mask ----
      const bool need_mask = (t0 + 31 > m0);
#pragma unroll
      for (int j = 0; j < 2; ++j)
#pragma unroll
        for (int r = 0; r < 8; ++r) {
          float x = lacc[j][r] * 0.125f;  // 1/sqrt(64)
          if (need_mask) {
            const int t = t0 + j * 16 + colb;
            const int m = m0 + r + half * 8;
            if (t > m) x = -1e30f;
          }
          lacc[j][r] = x;
        }
      // ---- online softmax (row stats per lane; reduce across 16-lane half) ----
      float tmax[8], tsum[8];
#pragma unroll
      for (int r = 0; r < 8; ++r) tmax[r] = fmaxf(lacc[0][r], lacc[1][r]);
#pragma unroll
      for (int off = 1; off < 16; off <<= 1)
#pragma unroll
        for (int r = 0; r < 8; ++r)
          tmax[r] = fmaxf(tmax[r], __shfl_xor(tmax[r], off, 16));
      float alpha[8];
#pragma unroll
      for (int r = 0; r < 8; ++r) {
        const float nm = fmaxf(mrow[r], tmax[r]);
        alpha[r] = __expf(mrow[r] - nm);
        mrow[r] = nm;
        tsum[r] = 0.f;
      }
#pragma unroll
      for (int j = 0; j < 2; ++j)
#pragma unroll
        for (int r = 0; r < 8; ++r) {
          const float p = __expf(lacc[j][r] - mrow[r]);
          lacc[j][r] = p;
          tsum[r] += p;
        }
#pragma unroll
      for (int off = 1; off < 16; off <<= 1)
#pragma unroll
        for (int r = 0; r < 8; ++r) tsum[r] += __shfl_xor(tsum[r], off, 16);
#pragma unroll
      for (int r = 0; r < 8; ++r) lrow[r] = lrow[r] * alpha[r] + tsum[r];
#pragma unroll
      for (int j = 0; j < 4; ++j)
#pragma unroll
        for (int r = 0; r < 8; ++r) acc[j][r] *= alpha[r];

      // ---- P (C layout) -> per-wave LDS -> A-fragment layout ----
#pragma unroll
      for (int j = 0; j < 2; ++j)
#pragma unroll
        for (int r = 0; r < 8; ++r)
          pbuf[wave][(r + half * 8) * PLD + j * 16 + colb] = (__bf16)lacc[j][r];

      v16bf pf;
      {
        const int row = lane & 15;
        const int sel = lane >> 4;
#pragma unroll
        for (int i = 0; i < 8; ++i) {
          pf[i]     = pbuf[wave][row * PLD + sel * 8 + i];
          pf[i + 8] = pbuf[wave][row * PLD + sel * 8 + 16 + i];
        }
      }
      // ---- O += P(16x32) @ V(32x64) (V from LDS, K-contiguous per hd) ----
#pragma unroll
      for (int j = 0; j < 4; ++j) {
        v16bf vf = load_b_frag(&vtile[(j * 16) * VLD], VLD, lane);
        acc[j] = wmma_bf16(pf, vf, acc[j]);
      }
    }
  }

  // ---- normalize and store merged-head bf16 activations [B,S,D] ----
  float inv[8];
#pragma unroll
  for (int r = 0; r < 8; ++r) inv[r] = 1.0f / lrow[r];
#pragma unroll
  for (int j = 0; j < 4; ++j)
#pragma unroll
    for (int r = 0; r < 8; ++r) {
      const int s   = m0 + r + half * 8;
      const int col = h * HDIM + j * 16 + colb;
      attnOut[((size_t)b * S_LEN + s) * DMODEL + col] = (__bf16)(acc[j][r] * inv[r]);
    }
}

extern "C" void kernel_launch(void* const* d_in, const int* in_sizes, int n_in,
                              void* d_out, int out_size, void* d_ws, size_t ws_size,
                              hipStream_t stream) {
  (void)in_sizes; (void)n_in; (void)out_size; (void)ws_size;
  const float* q  = (const float*)d_in[0];
  const float* k  = (const float*)d_in[1];
  const float* v  = (const float*)d_in[2];
  // d_in[3] = causal mask: recomputed analytically on device, not read.
  const float* wq = (const float*)d_in[4];
  const float* wk = (const float*)d_in[5];
  const float* wv = (const float*)d_in[6];
  const float* wo = (const float*)d_in[7];

  const size_t ACT = (size_t)BATCH * S_LEN * DMODEL;  // 4 Mi elems
  const size_t WEL = (size_t)DMODEL * DMODEL;         // 1 Mi elems

  char* ws = (char*)d_ws;
  __bf16* qb    = (__bf16*)ws; ws += ACT * 2;
  __bf16* kb    = (__bf16*)ws; ws += ACT * 2;
  __bf16* vb    = (__bf16*)ws; ws += ACT * 2;
  __bf16* wqb   = (__bf16*)ws; ws += WEL * 2;
  __bf16* wkb   = (__bf16*)ws; ws += WEL * 2;
  __bf16* wvb   = (__bf16*)ws; ws += WEL * 2;
  __bf16* wob   = (__bf16*)ws; ws += WEL * 2;
  __bf16* Qh    = (__bf16*)ws; ws += ACT * 2;
  __bf16* Kh    = (__bf16*)ws; ws += ACT * 2;
  __bf16* Vt    = (__bf16*)ws; ws += ACT * 2;
  __bf16* attnb = (__bf16*)ws; ws += ACT * 2;         // ~64 MiB total

  const int tb = 256;  // 8 waves (wave32)
  cast_f32_bf16<<<(int)((ACT + tb - 1) / tb), tb, 0, stream>>>(q,  qb,  (int)ACT);
  cast_f32_bf16<<<(int)((ACT + tb - 1) / tb), tb, 0, stream>>>(k,  kb,  (int)ACT);
  cast_f32_bf16<<<(int)((ACT + tb - 1) / tb), tb, 0, stream>>>(v,  vb,  (int)ACT);
  cast_f32_bf16<<<(int)((WEL + tb - 1) / tb), tb, 0, stream>>>(wq, wqb, (int)WEL);
  cast_f32_bf16<<<(int)((WEL + tb - 1) / tb), tb, 0, stream>>>(wk, wkb, (int)WEL);
  cast_f32_bf16<<<(int)((WEL + tb - 1) / tb), tb, 0, stream>>>(wv, wvb, (int)WEL);
  cast_f32_bf16<<<(int)((WEL + tb - 1) / tb), tb, 0, stream>>>(wo, wob, (int)WEL);

  dim3 gg(DMODEL / 128, (unsigned)((BATCH * S_LEN) / 128));  // (8, 32)
  gemm_xWT<0><<<gg, tb, 0, stream>>>(qb, wqb, Qh);
  gemm_xWT<0><<<gg, tb, 0, stream>>>(kb, wkb, Kh);
  gemm_xWT<1><<<gg, tb, 0, stream>>>(vb, wvb, Vt);

  flash_attn<<<dim3(S_LEN / 128, BATCH * NHEAD), tb, 0, stream>>>(Qh, Kh, Vt, attnb);

  gemm_xWT<2><<<gg, tb, 0, stream>>>(attnb, wob, d_out);
}
